// LiquidSelfAttention_14199161880581
// MI455X (gfx1250) — compile-verified, matrix operand order checked by reading, not run on hardware
//
#include <hip/hip_runtime.h>
#include <hip/hip_bf16.h>
#include <math.h>

// ---------------------------------------------------------------------------
// Problem constants (from reference)
// ---------------------------------------------------------------------------
#define B_SZ   4
#define T_SZ   4096
#define D_SZ   1024
#define H_SZ   16
#define HD_SZ  64
#define M_TOT  (B_SZ * T_SZ)      // 16384 rows
#define N_CAT  (4 * D_SZ)         // 4096 fused output columns (Q|K|V|G)
#define K_DIM  D_SZ               // 1024

#define BM 128
#define BN 128
#define BK 64                     // two WMMA K-steps per LDS stage
#define NKT (K_DIM / BK)          // 16 staged tiles

typedef __attribute__((ext_vector_type(16))) __bf16 v16bf;
typedef __attribute__((ext_vector_type(8)))  float  v8f;

// round-to-nearest-even fp32 -> bf16 (bit pattern in ushort)
__device__ __forceinline__ unsigned short f32_to_bf16(float f) {
    unsigned int u = __float_as_uint(f);
    unsigned int r = u + 0x7FFFu + ((u >> 16) & 1u);
    return (unsigned short)(r >> 16);
}

// CDNA5 async DMA: global -> LDS, 16 bytes, tracked by ASYNCcnt.
// Flat-address low 32 bits of a __shared__ pointer are the LDS byte offset.
__device__ __forceinline__ void async_copy_b128(void* lds_ptr, const void* gptr) {
    unsigned int  la = (unsigned int)(size_t)lds_ptr;
    unsigned long long ga = (unsigned long long)(size_t)gptr;
    asm volatile("global_load_async_to_lds_b128 %0, %1, off"
                 :: "v"(la), "v"(ga) : "memory");
}
__device__ __forceinline__ void wait_asynccnt0() {
    asm volatile("s_wait_asynccnt 0" ::: "memory");
}

// ---------------------------------------------------------------------------
// K0: x (fp32) -> bf16, vectorized 4-wide
// ---------------------------------------------------------------------------
__global__ __launch_bounds__(256) void cvt_x_kernel(const float4* __restrict__ x4,
                                                    ushort4* __restrict__ xb4) {
    size_t i = (size_t)blockIdx.x * 256 + threadIdx.x;   // 4,194,304 vec4 elems
    float4 f = x4[i];
    ushort4 o;
    o.x = f32_to_bf16(f.x);
    o.y = f32_to_bf16(f.y);
    o.z = f32_to_bf16(f.z);
    o.w = f32_to_bf16(f.w);
    xb4[i] = o;
}

// ---------------------------------------------------------------------------
// K1: pack {Wq,Wk,Wv,Wg} (fp32) -> one [4096,1024] bf16 matrix
// ---------------------------------------------------------------------------
__global__ __launch_bounds__(256) void cvt_w_kernel(const float* __restrict__ Wq,
                                                    const float* __restrict__ Wk,
                                                    const float* __restrict__ Wv,
                                                    const float* __restrict__ Wg,
                                                    unsigned short* __restrict__ Wbf) {
    size_t idx = (size_t)blockIdx.x * 256 + threadIdx.x; // 4,194,304 elems
    int col   = (int)(idx & 1023);
    int row   = (int)(idx >> 10);     // 0..4095
    int which = row >> 10;            // 0..3
    int rloc  = row & 1023;
    const float* src = (which == 0) ? Wq : (which == 1) ? Wk : (which == 2) ? Wv : Wg;
    Wbf[idx] = f32_to_bf16(src[(size_t)rloc * 1024 + col]);
}

// ---------------------------------------------------------------------------
// K2: zero the k_sum / kv_sum accumulators (contiguous 8192 floats)
// ---------------------------------------------------------------------------
__global__ __launch_bounds__(256) void zero_kernel(float* __restrict__ p) {
    p[(size_t)blockIdx.x * 256 + threadIdx.x] = 0.0f;
}

// ---------------------------------------------------------------------------
// K3: fused WMMA GEMM  Y[m,n] = sum_k X[m,k] * Wcat[n,k]   (bf16 -> f32 acc)
//     double-buffered LDS tiles filled by async global->LDS DMA, overlapped
//     with v_wmma_f32_16x16x32_bf16 compute.
//     epilogue per 1024-column segment:
//       seg 0: Q = elu(y)+1      seg 1: K = elu(y)+1
//       seg 2: V = y             seg 3: G = sigmoid(y + bg)
// ---------------------------------------------------------------------------
__global__ __launch_bounds__(256) void gemm_qkvg_kernel(
        const unsigned short* __restrict__ X,    // [M_TOT, 1024] bf16 bits
        const unsigned short* __restrict__ W,    // [4096, 1024] bf16 bits
        const float* __restrict__ bg,            // [1024]
        float* __restrict__ Qo, float* __restrict__ Ko,
        float* __restrict__ Vo, float* __restrict__ Go) {
    __shared__ alignas(16) unsigned short sA[2][BM][BK];   // 32 KB
    __shared__ alignas(16) unsigned short sB[2][BN][BK];   // 32 KB

    const int tid  = threadIdx.x;
    const int wid  = tid >> 5;            // 8 waves (wave32)
    const int lane = tid & 31;
    const int bm   = blockIdx.x * BM;
    const int bn   = blockIdx.y * BN;

    const int wm = (wid >> 1) * 32;       // wave row offset: 0/32/64/96
    const int wn = (wid & 1) * 64;        // wave col offset: 0/64

    const int fl = lane & 15;             // fragment row within 16-tile
    const int fk = (lane >> 4) * 16;      // fragment K half: 0 or 16

    // staging assignment: 1024 16B-chunks per tile, 4 per thread
    const int srow[4] = { (tid + 0) >> 3, (tid + 256) >> 3,
                          (tid + 512) >> 3, (tid + 768) >> 3 };
    const int scol = (tid & 7) * 8;       // bf16 column of 16B chunk

    // ---- prologue: async-fill buffer 0 with K-tile 0 ----
    #pragma unroll
    for (int i = 0; i < 4; ++i) {
        async_copy_b128(&sA[0][srow[i]][scol],
                        &X[(size_t)(bm + srow[i]) * K_DIM + scol]);
        async_copy_b128(&sB[0][srow[i]][scol],
                        &W[(size_t)(bn + srow[i]) * K_DIM + scol]);
    }

    v8f acc[2][4] = {};

    for (int kt = 0; kt < NKT; ++kt) {
        const int buf = kt & 1;
        // make tile kt visible to all waves; also guarantees the other
        // buffer's readers (iteration kt-1) are done before we refill it
        wait_asynccnt0();
        __syncthreads();

        // ---- kick off async fill of the other buffer for tile kt+1 ----
        if (kt + 1 < NKT) {
            const int nb = buf ^ 1;
            const int kcol = (kt + 1) * BK;
            #pragma unroll
            for (int i = 0; i < 4; ++i) {
                async_copy_b128(&sA[nb][srow[i]][scol],
                                &X[(size_t)(bm + srow[i]) * K_DIM + kcol + scol]);
                async_copy_b128(&sB[nb][srow[i]][scol],
                                &W[(size_t)(bn + srow[i]) * K_DIM + kcol + scol]);
            }
            // warm L2 two tiles ahead
            if (kt + 2 < NKT)
                __builtin_prefetch(&X[(size_t)(bm + srow[0]) * K_DIM + (kt + 2) * BK], 0, 1);
        }

        // ---- compute: 2 K-steps x (2x4) WMMA tiles per wave ----
        #pragma unroll
        for (int ks = 0; ks < 2; ++ks) {
            v16bf afrag[2];
            #pragma unroll
            for (int i = 0; i < 2; ++i)
                afrag[i] = *reinterpret_cast<const v16bf*>(
                    &sA[buf][wm + i * 16 + fl][ks * 32 + fk]);
            v16bf bfrag[4];
            #pragma unroll
            for (int j = 0; j < 4; ++j)
                bfrag[j] = *reinterpret_cast<const v16bf*>(
                    &sB[buf][wn + j * 16 + fl][ks * 32 + fk]);
            #pragma unroll
            for (int i = 0; i < 2; ++i)
                #pragma unroll
                for (int j = 0; j < 4; ++j)
                    acc[i][j] = __builtin_amdgcn_wmma_f32_16x16x32_bf16(
                        false, afrag[i], false, bfrag[j],
                        (short)0, acc[i][j], false, false);
        }
    }

    // ---- epilogue: C layout => col = lane&15, row = (lane>>4)*8 + r ----
    #pragma unroll
    for (int j = 0; j < 4; ++j) {
        int gn16  = bn + wn + j * 16;           // tile stays inside one segment
        int which = gn16 >> 10;                 // 0=Q 1=K 2=V 3=G (wave-uniform)
        int nloc  = (gn16 & 1023) + (lane & 15);
        float* dst  = (which == 0) ? Qo : (which == 1) ? Ko : (which == 2) ? Vo : Go;
        float  badd = (which == 3) ? bg[nloc] : 0.0f;
        #pragma unroll
        for (int i = 0; i < 2; ++i) {
            int gmBase = bm + wm + i * 16 + ((lane >> 4) * 8);
            #pragma unroll
            for (int r = 0; r < 8; ++r) {
                float v = acc[i][j][r];
                if (which <= 1)       v = (v > 0.0f) ? (v + 1.0f) : __expf(v); // elu+1
                else if (which == 3)  v = 1.0f / (1.0f + __expf(-(v + badd))); // sigmoid
                dst[(size_t)(gmBase + r) * D_SZ + nloc] = v;
            }
        }
    }
}

// ---------------------------------------------------------------------------
// K4: k_sum[b,c] = sum_t K[b,t,c];  kv_sum[b,c] = sum_t K[b,t,c]*V[b,t,c]
//     grid: b(4) x dtile(4, 256 cols) x tchunk(16, 256 t) ; atomic finish
// ---------------------------------------------------------------------------
__global__ __launch_bounds__(256) void reduce_sums_kernel(
        const float* __restrict__ Kf, const float* __restrict__ Vf,
        float* __restrict__ ksum, float* __restrict__ kvsum) {
    int bidx  = blockIdx.x;           // 0..255
    int b     = bidx >> 6;
    int dtile = (bidx >> 4) & 3;
    int tch   = bidx & 15;
    int col   = dtile * 256 + threadIdx.x;
    size_t base = ((size_t)(b * T_SZ + tch * 256)) * D_SZ + col;
    float sk = 0.0f, skv = 0.0f;
    for (int t = 0; t < 256; ++t) {
        float kv = Kf[base + (size_t)t * D_SZ];
        float vv = Vf[base + (size_t)t * D_SZ];
        sk  += kv;
        skv += kv * vv;
    }
    atomicAdd(&ksum[b * D_SZ + col], sk);
    atomicAdd(&kvsum[b * D_SZ + col], skv);
}

// ---------------------------------------------------------------------------
// K5: per (b,t,h): s = q . kv_sum ; z = q . k_sum + eps ; o = s/z
//     out = g*o + (1-g)*x    (o broadcast over head_dim)
//     one wave32 per (b,t,h); 2 dims per lane + xor-shuffle reduce
// ---------------------------------------------------------------------------
__global__ __launch_bounds__(256) void final_kernel(
        const float* __restrict__ Qf, const float* __restrict__ Gf,
        const float* __restrict__ ksum, const float* __restrict__ kvsum,
        const float* __restrict__ x, float* __restrict__ out) {
    int gw   = blockIdx.x * 8 + (threadIdx.x >> 5);  // 0 .. B*T*H-1
    int lane = threadIdx.x & 31;
    int h  = gw & 15;
    int bt = gw >> 4;                                // b*T + t
    int b  = bt >> 12;
    size_t rowOff = (size_t)bt * D_SZ + h * HD_SZ;
    int d0 = lane, d1 = lane + 32;

    float q0 = Qf[rowOff + d0], q1 = Qf[rowOff + d1];
    const float* kv = kvsum + b * D_SZ + h * HD_SZ;
    const float* ks = ksum  + b * D_SZ + h * HD_SZ;
    float s = q0 * kv[d0] + q1 * kv[d1];
    float z = q0 * ks[d0] + q1 * ks[d1];
    #pragma unroll
    for (int off = 16; off > 0; off >>= 1) {
        s += __shfl_xor(s, off, 32);
        z += __shfl_xor(z, off, 32);
    }
    float o = s / (z + 1e-6f);

    float g0 = Gf[rowOff + d0], g1 = Gf[rowOff + d1];
    float x0 = x[rowOff + d0],  x1 = x[rowOff + d1];
    out[rowOff + d0] = g0 * o + (1.0f - g0) * x0;
    out[rowOff + d1] = g1 * o + (1.0f - g1) * x1;
}

// ---------------------------------------------------------------------------
// host launcher
// ---------------------------------------------------------------------------
extern "C" void kernel_launch(void* const* d_in, const int* in_sizes, int n_in,
                              void* d_out, int out_size, void* d_ws, size_t ws_size,
                              hipStream_t stream) {
    const float* x  = (const float*)d_in[0];
    const float* Wq = (const float*)d_in[1];
    const float* Wk = (const float*)d_in[2];
    const float* Wv = (const float*)d_in[3];
    // d_in[4] (Wo) is unused by the reference
    const float* Wg = (const float*)d_in[5];
    const float* bg = (const float*)d_in[6];
    float* out = (float*)d_out;

    // workspace carve-up (all offsets 16B-aligned)
    char* ws = (char*)d_ws;
    unsigned short* xbf = (unsigned short*)ws;  ws += (size_t)M_TOT * K_DIM * 2;   // 32 MB
    unsigned short* Wbf = (unsigned short*)ws;  ws += (size_t)N_CAT * K_DIM * 2;   //  8 MB
    float* Qf = (float*)ws;  ws += (size_t)M_TOT * D_SZ * 4;                       // 64 MB
    float* Kf = (float*)ws;  ws += (size_t)M_TOT * D_SZ * 4;                       // 64 MB
    float* Vf = (float*)ws;  ws += (size_t)M_TOT * D_SZ * 4;                       // 64 MB
    float* Gf = (float*)ws;  ws += (size_t)M_TOT * D_SZ * 4;                       // 64 MB
    float* ksum  = (float*)ws;  ws += (size_t)B_SZ * D_SZ * 4;                     // 16 KB
    float* kvsum = (float*)ws;  ws += (size_t)B_SZ * D_SZ * 4;                     // 16 KB

    // K0: x -> bf16 (16,777,216 elems as 4,194,304 float4)
    cvt_x_kernel<<<16384, 256, 0, stream>>>((const float4*)x, (ushort4*)xbf);

    // K1: weights -> packed bf16 [4096,1024]
    cvt_w_kernel<<<16384, 256, 0, stream>>>(Wq, Wk, Wv, Wg, Wbf);

    // K2: zero ksum+kvsum (contiguous 8192 floats)
    zero_kernel<<<32, 256, 0, stream>>>(ksum);

    // K3: fused WMMA GEMM -> Q,K,V,G (async double-buffered)
    dim3 ggrid(M_TOT / BM, N_CAT / BN);    // 128 x 32
    gemm_qkvg_kernel<<<ggrid, 256, 0, stream>>>(xbf, Wbf, bg, Qf, Kf, Vf, Gf);

    // K4: T-reductions
    reduce_sums_kernel<<<256, 256, 0, stream>>>(Kf, Vf, ksum, kvsum);

    // K5: attention scalar + gated blend
    final_kernel<<<(B_SZ * T_SZ * H_SZ) / 8, 256, 0, stream>>>(Qf, Gf, ksum, kvsum, x, out);
}
